// WindowAttention_24043226923013
// MI455X (gfx1250) — matrix operand order repaired and checked by reading, not compiled, not run on hardware
//
#include <hip/hip_runtime.h>

#define N_TOKS 343
#define NP     352          // 22 tiles of 16 (also padded BM stride)
#define DIMC   96
#define HEADS  3
#define HD     32
#define NWIN   64
#define NBATCH 512
#define LOG2E  1.4426950408889634f
#define QK_SCALE (0.17677669529663687f * LOG2E)   // 1/sqrt(32) * log2(e)
#define BM_PAD (-60000.0f)                        // f16-representable -inf proxy

// workspace layout (bytes)
#define OFF_WH   0u
#define OFF_PWH  55296u
#define OFF_TAB  73728u                 // B3 (f32, fallback) or BM (f16 padded, combined)
#define OFF_OB_C 47652864u              // 73728 + 192*352*352*2
#define OFF_OB_F 1485568u               // 73728 + 3*343*343*4, aligned
#define OB_BYTES 34603008u              // 512*352*96*2
#define NEED_C   (OFF_OB_C + OB_BYTES)  // ~82.3 MB

typedef __attribute__((ext_vector_type(16))) _Float16 v16h;
typedef __attribute__((ext_vector_type(8)))  _Float16 v8h;
typedef __attribute__((ext_vector_type(8)))  float    v8f;

__device__ __forceinline__ v16h cat8(v8h a, v8h b) {
    return __builtin_shufflevector(a, b, 0,1,2,3,4,5,6,7,8,9,10,11,12,13,14,15);
}
// B-matrix 32x16 f16 fragment; p = 32 K-values of this lane's column; elem i = p[(i&7)+16*(i>>3)+8*half]
__device__ __forceinline__ v16h load_b_f16(const _Float16* p, int half) {
    const v8h* lo = (const v8h*)(p + 8 * half);
    const v8h* hi = (const v8h*)(p + 16 + 8 * half);
    return cat8(lo[0], hi[0]);
}
// A-matrix 16x32 f16 fragment; p = 32 K-values of this lane's row; elem i = p[(i&7)+16*(i>>3)+8*half]
__device__ __forceinline__ v16h load_a_f16(const _Float16* p, int half) {
    const v8h* lo = (const v8h*)(p + 8 * half);
    const v8h* hi = (const v8h*)(p + 16 + 8 * half);
    return cat8(lo[0], hi[0]);
}
__device__ __forceinline__ v8f wmma_f16(v16h a, v16h b, v8f c) {
    return __builtin_amdgcn_wmma_f32_16x16x32_f16(false, a, false, b, (short)0, c, false, false);
}

// ---------------------------------------------------------------------------
// Prep kernels
// ---------------------------------------------------------------------------
__global__ void prep_w_kernel(const float* __restrict__ qkv_w, const float* __restrict__ proj_w,
                              _Float16* __restrict__ Wh, _Float16* __restrict__ Pwh) {
    const int i = blockIdx.x * blockDim.x + threadIdx.x;
    if (i < 3 * DIMC * DIMC) Wh[i]  = (_Float16)qkv_w[i];   // native [n][k]
    if (i < DIMC * DIMC)     Pwh[i] = (_Float16)proj_w[i];
}

// fallback: bias only, f32 [3][343][343]
__global__ void prep_b3_kernel(const float* __restrict__ rpb, const int* __restrict__ relidx,
                               float* __restrict__ B3) {
    const int i = blockIdx.x * blockDim.x + threadIdx.x;
    if (i < HEADS * N_TOKS * N_TOKS) {
        const int hh = i / (N_TOKS * N_TOKS), r = i % (N_TOKS * N_TOKS);
        B3[i] = rpb[relidx[r] * HEADS + hh];
    }
}

// combined: BM[w][h][352][352] = (bias + mask) * log2e, f16; pad cells = -60000
__global__ void prep_bm_kernel(const float* __restrict__ rpb, const int* __restrict__ relidx,
                               const float* __restrict__ maskg, _Float16* __restrict__ BM) {
    const int i = blockIdx.x * blockDim.x + threadIdx.x;
    const int NN = NP * NP;
    if (i < NWIN * HEADS * NN) {
        const int wdx  = i / (HEADS * NN);
        const int hh   = (i / NN) % HEADS;
        const int rest = i % NN;
        const int n = rest / NP, m = rest % NP;
        float v = BM_PAD;
        if (n < N_TOKS && m < N_TOKS)
            v = (rpb[relidx[n * N_TOKS + m] * HEADS + hh] +
                 maskg[(size_t)wdx * N_TOKS * N_TOKS + n * N_TOKS + m]) * LOG2E;
        BM[i] = (_Float16)v;
    }
}

// ---------------------------------------------------------------------------
// Kernel 1: fused QKV-projection + flash attention, per (window b, head h)
// ---------------------------------------------------------------------------
template <bool COMBINED>
__global__ __launch_bounds__(256)
void attn_kernel(const float* __restrict__ x, const float* __restrict__ maskg,
                 const float* __restrict__ qkvb, const _Float16* __restrict__ Wh,
                 const float* __restrict__ bias3, const _Float16* __restrict__ BM,
                 _Float16* __restrict__ Obuf) {
    __shared__ alignas(32) _Float16 Ks[NP][HD];        // K row-major   (22.0 KB)
    __shared__ alignas(32) _Float16 VTs[HD][NP];       // V transposed  (22.0 KB)
    __shared__ alignas(32) _Float16 Ps[8][16][32];     // per-wave D->A bounce (8 KB)

    const int b = blockIdx.x, h = blockIdx.y;
    const int w = threadIdx.x >> 5, lane = threadIdx.x & 31;
    const int l15 = lane & 15, half = lane >> 4;

    v16h qa[3];   // Q A-fragments for this wave's M tiles

    // -------- Stage 1: QKV = x @ Wqkv + b (head slice), WMMA --------
    #pragma unroll
    for (int mi = 0; mi < 3; ++mi) {
        const int mt = w + 8 * mi;
        if (mt < 22) {
            const int m0 = mt * 16;
            int xrow = m0 + l15; if (xrow > N_TOKS - 1) xrow = N_TOKS - 1;
            const float* xr = x + ((size_t)b * N_TOKS + xrow) * DIMC;
            v16h xa[3];
            #pragma unroll
            for (int ks = 0; ks < 3; ++ks) {
                const float* bp = xr + ks * 32 + 8 * half;
                #pragma unroll
                for (int i = 0; i < 8; ++i) {
                    xa[ks][i]     = (_Float16)bp[i];
                    xa[ks][i + 8] = (_Float16)bp[16 + i];
                }
            }
            #pragma unroll
            for (int sec = 0; sec < 3; ++sec) {        // 0=q 1=k 2=v
                #pragma unroll
                for (int nt = 0; nt < 2; ++nt) {
                    const int ncol = sec * DIMC + h * HD + nt * 16 + l15;
                    const _Float16* wp = Wh + (size_t)ncol * DIMC;   // column ncol, K-contig
                    v8f acc = {0.f,0.f,0.f,0.f,0.f,0.f,0.f,0.f};
                    #pragma unroll
                    for (int ks = 0; ks < 3; ++ks) {
                        v16h wb = load_b_f16(wp + ks * 32, half);
                        acc = wmma_f16(xa[ks], wb, acc);
                    }
                    const float bv = qkvb[ncol];
                    #pragma unroll
                    for (int r = 0; r < 8; ++r) {
                        const float v = acc[r] + bv;
                        const int row = m0 + r + 8 * half;
                        if (sec == 0)
                            Ps[w][r + 8 * half][nt * 16 + l15] = (_Float16)(v * QK_SCALE);
                        else if (sec == 1)
                            Ks[row][nt * 16 + l15] = (_Float16)v;
                        else
                            VTs[nt * 16 + l15][row] = (_Float16)v;
                    }
                }
                if (sec == 0)          // D-layout -> A-layout via LDS bounce
                    qa[mi] = load_a_f16(&Ps[w][l15][0], half);
            }
        }
    }
    __syncthreads();

    // -------- Stage 2: flash attention over 11 chunks of 32 keys --------
    float mst[3][8];
    v8f o0[3], o1[3], ol[3];           // ol = running row-sum via all-ones WMMA
    #pragma unroll
    for (int mi = 0; mi < 3; ++mi) {
        #pragma unroll
        for (int r = 0; r < 8; ++r) {
            mst[mi][r] = -1e30f;
            o0[mi][r] = 0.f; o1[mi][r] = 0.f; ol[mi][r] = 0.f;
        }
    }
    v16h ones;
    #pragma unroll
    for (int i = 0; i < 16; ++i) ones[i] = (_Float16)1.0f;

    const float* maskw = maskg + (size_t)(b & (NWIN - 1)) * N_TOKS * N_TOKS;
    const float* biash = bias3 + (size_t)h * N_TOKS * N_TOKS;
    const _Float16* bmp = BM + ((size_t)(b & (NWIN - 1)) * HEADS + h) * NP * NP;

    for (int c = 0; c < 11; ++c) {
        const int col0 = c * 32;
        v16h kb0 = load_b_f16(&Ks[col0 + l15][0], half);        // B[k][n] = K[n][k]
        v16h kb1 = load_b_f16(&Ks[col0 + 16 + l15][0], half);
        v16h vb0 = load_b_f16(&VTs[l15][col0], half);           // B[k][n] = V[k][n]
        v16h vb1 = load_b_f16(&VTs[16 + l15][col0], half);

        const int gc0 = col0 + l15, gc1 = gc0 + 16;

        #pragma unroll
        for (int mi = 0; mi < 3; ++mi) {
            const int mt = w + 8 * mi;
            if (mt < 22) {
                const int m0 = mt * 16;
                v8f z = {0.f,0.f,0.f,0.f,0.f,0.f,0.f,0.f};
                v8f s0 = wmma_f16(qa[mi], kb0, z);
                v8f s1 = wmma_f16(qa[mi], kb1, z);
                float cm[8], p0[8], p1[8];
                if (COMBINED) {
                    // branch-free: pad cells of BM carry the -inf sentinel
                    const _Float16* bmr = bmp + (m0 + 8 * half) * NP;
                    #pragma unroll
                    for (int r = 0; r < 8; ++r) {
                        const float a0 = s0[r] + (float)bmr[r * NP + gc0];
                        const float a1 = s1[r] + (float)bmr[r * NP + gc1];
                        s0[r] = a0; s1[r] = a1;
                        cm[r] = fmaxf(a0, a1);
                    }
                } else {
                    const bool cv0 = gc0 < N_TOKS, cv1 = gc1 < N_TOKS;
                    #pragma unroll
                    for (int r = 0; r < 8; ++r) {
                        const int row = m0 + r + 8 * half;
                        float a0 = s0[r], a1 = s1[r];
                        if (row < N_TOKS) {
                            if (cv0) a0 += (biash[row * N_TOKS + gc0] + maskw[row * N_TOKS + gc0]) * LOG2E;
                            if (cv1) a1 += (biash[row * N_TOKS + gc1] + maskw[row * N_TOKS + gc1]) * LOG2E;
                        }
                        if (!cv0) a0 = -1e30f;
                        if (!cv1) a1 = -1e30f;
                        s0[r] = a0; s1[r] = a1;
                        cm[r] = fmaxf(a0, a1);
                    }
                }
                // row-max across the 16 lanes of each half (wave32-safe)
                #pragma unroll
                for (int r = 0; r < 8; ++r) {
                    #pragma unroll
                    for (int m = 1; m < 16; m <<= 1)
                        cm[r] = fmaxf(cm[r], __shfl_xor(cm[r], m, 32));
                }
                #pragma unroll
                for (int r = 0; r < 8; ++r) {
                    const float nm  = fmaxf(mst[mi][r], cm[r]);
                    const float fac = __builtin_exp2f(mst[mi][r] - nm);   // log2 domain
                    p0[r] = __builtin_exp2f(s0[r] - nm);
                    p1[r] = __builtin_exp2f(s1[r] - nm);
                    mst[mi][r] = nm;
                    o0[mi][r] *= fac;
                    o1[mi][r] *= fac;
                    ol[mi][r] *= fac;
                }
                // P: D-layout -> A-layout via per-wave LDS bounce
                #pragma unroll
                for (int r = 0; r < 8; ++r) {
                    Ps[w][r + 8 * half][l15]      = (_Float16)p0[r];
                    Ps[w][r + 8 * half][16 + l15] = (_Float16)p1[r];
                }
                v16h pa = load_a_f16(&Ps[w][l15][0], half);
                o0[mi] = wmma_f16(pa, vb0, o0[mi]);
                o1[mi] = wmma_f16(pa, vb1, o1[mi]);
                ol[mi] = wmma_f16(pa, ones, ol[mi]);   // row-sum recurrence
            }
        }
    }

    // -------- finalize: O /= l, store f16 head slice --------
    #pragma unroll
    for (int mi = 0; mi < 3; ++mi) {
        const int mt = w + 8 * mi;
        if (mt < 22) {
            const int m0 = mt * 16;
            #pragma unroll
            for (int r = 0; r < 8; ++r) {
                const int row = m0 + r + 8 * half;
                const float inv = 1.0f / ol[mi][r];
                _Float16* op = Obuf + ((size_t)b * NP + row) * DIMC + h * HD;
                op[l15]      = (_Float16)(o0[mi][r] * inv);
                op[16 + l15] = (_Float16)(o1[mi][r] * inv);
            }
        }
    }
}

// ---------------------------------------------------------------------------
// Kernel 2: output projection  out = O @ proj_w.T + proj_b  (WMMA)
// ---------------------------------------------------------------------------
__global__ __launch_bounds__(256)
void proj_kernel(const _Float16* __restrict__ Obuf, const _Float16* __restrict__ Pwh,
                 const float* __restrict__ projb, float* __restrict__ out) {
    const int b = blockIdx.x;
    const int w = threadIdx.x >> 5, lane = threadIdx.x & 31;
    const int l15 = lane & 15, half = lane >> 4;

    for (int t = w; t < 22 * 6; t += 8) {
        const int mt = t / 6, nt = t % 6;
        const int m0 = mt * 16, n0 = nt * 16;
        const _Float16* ar = Obuf + ((size_t)b * NP + m0 + l15) * DIMC;   // A row, K-contig
        const _Float16* bp = Pwh + (size_t)(n0 + l15) * DIMC;             // B col, K-contig
        v8f acc = {0.f,0.f,0.f,0.f,0.f,0.f,0.f,0.f};
        #pragma unroll
        for (int ks = 0; ks < 3; ++ks) {
            v16h a  = load_a_f16(ar + ks * 32, half);
            v16h bb = load_b_f16(bp + ks * 32, half);
            acc = wmma_f16(a, bb, acc);
        }
        const float pb = projb[n0 + l15];
        float* orow = out + ((size_t)b * N_TOKS + m0 + 8 * half) * DIMC + n0 + l15;
        if (m0 + 16 <= N_TOKS) {
            // full tile: unconditional stores
            #pragma unroll
            for (int r = 0; r < 8; ++r)
                orow[r * DIMC] = acc[r] + pb;
        } else {
            #pragma unroll
            for (int r = 0; r < 8; ++r) {
                const int row = m0 + r + 8 * half;
                if (row < N_TOKS)
                    out[((size_t)b * N_TOKS + row) * DIMC + n0 + l15] = acc[r] + pb;
            }
        }
    }
}

extern "C" void kernel_launch(void* const* d_in, const int* in_sizes, int n_in,
                              void* d_out, int out_size, void* d_ws, size_t ws_size,
                              hipStream_t stream) {
    const float* x      = (const float*)d_in[0];
    const float* mask   = (const float*)d_in[1];
    const float* qkv_w  = (const float*)d_in[2];
    const float* qkv_b  = (const float*)d_in[3];
    const float* proj_w = (const float*)d_in[4];
    const float* proj_b = (const float*)d_in[5];
    const float* rpb    = (const float*)d_in[6];
    const int*   relidx = (const int*)d_in[7];

    char* ws = (char*)d_ws;
    _Float16* Wh  = (_Float16*)(ws + OFF_WH);
    _Float16* Pwh = (_Float16*)(ws + OFF_PWH);
    float* out = (float*)d_out;

    prep_w_kernel<<<(3 * DIMC * DIMC + 255) / 256, 256, 0, stream>>>(qkv_w, proj_w, Wh, Pwh);

    const bool combined = ws_size >= (size_t)NEED_C;
    if (combined) {
        _Float16* BM = (_Float16*)(ws + OFF_TAB);
        _Float16* Ob = (_Float16*)(ws + OFF_OB_C);
        const int nbm = NWIN * HEADS * NP * NP;
        prep_bm_kernel<<<(nbm + 255) / 256, 256, 0, stream>>>(rpb, relidx, mask, BM);
        attn_kernel<true><<<dim3(NBATCH, HEADS), 256, 0, stream>>>(
            x, mask, qkv_b, Wh, (const float*)nullptr, BM, Ob);
        proj_kernel<<<NBATCH, 256, 0, stream>>>(Ob, Pwh, proj_b, out);
    } else {
        float*    B3 = (float*)(ws + OFF_TAB);
        _Float16* Ob = (_Float16*)(ws + OFF_OB_F);
        const int nb3 = HEADS * N_TOKS * N_TOKS;
        prep_b3_kernel<<<(nb3 + 255) / 256, 256, 0, stream>>>(rpb, relidx, B3);
        attn_kernel<false><<<dim3(NBATCH, HEADS), 256, 0, stream>>>(
            x, mask, qkv_b, Wh, B3, (const _Float16*)nullptr, Ob);
        proj_kernel<<<NBATCH, 256, 0, stream>>>(Ob, Pwh, proj_b, out);
    }
}
